// CMAFM_75033078661474
// MI455X (gfx1250) — compile-verified, matrix operand order checked by reading, not run on hardware
//
#include <hip/hip_runtime.h>

typedef __attribute__((ext_vector_type(16))) _Float16 v16h;
typedef __attribute__((ext_vector_type(8)))  float    v8f;
typedef __attribute__((ext_vector_type(4)))  float    v4f;

#define NB 8          // batches
#define LCOLS 128
#define CCOLS 128
#define XCOLS 256     // L + C
#define OUTC 256
#define HDIM 170
#define CADIM 512
#define WAVES_PER_BLK 8
#define ROWS_PER_BLK (WAVES_PER_BLK * 16)

// ---------------------------------------------------------------- prep ----
// wt layout (f16): wt1[n*256+k] = Wf1[k*256+n], wt2 follows (n-major, K contiguous)
__global__ __launch_bounds__(256) void CMAFM_prep(const float* __restrict__ Wf1,
                                                  const float* __restrict__ Wf2,
                                                  _Float16* __restrict__ wt,
                                                  float* __restrict__ gmin,
                                                  float* __restrict__ gmax) {
    int i = blockIdx.x * 256 + threadIdx.x;
    if (i < 65536) {
        int n = i >> 8, k = i & 255;
        wt[i] = (_Float16)Wf1[k * 256 + n];
    } else if (i < 131072) {
        int j = i - 65536;
        int n = j >> 8, k = j & 255;
        wt[i] = (_Float16)Wf2[k * 256 + n];
    } else if (i < 131072 + 2048) {
        gmin[i - 131072] = __builtin_huge_valf();
    } else if (i < 131072 + 4096) {
        gmax[i - 131072 - 2048] = -__builtin_huge_valf();
    }
}

// ----------------------------------------------------- segment min/max ----
__device__ __forceinline__ void atomicMinF(float* addr, float v) {
    if (!(v < 0.0f))
        atomicMin((int*)addr, __float_as_int(v));
    else
        atomicMax((unsigned int*)addr, __float_as_uint(v));
}
__device__ __forceinline__ void atomicMaxF(float* addr, float v) {
    if (!(v < 0.0f))
        atomicMax((int*)addr, __float_as_int(v));
    else
        atomicMin((unsigned int*)addr, __float_as_uint(v));
}

__global__ __launch_bounds__(256) void CMAFM_segminmax(const float* __restrict__ lidar,
                                                       const float* __restrict__ cam,
                                                       const int* __restrict__ bidx,
                                                       float* __restrict__ gmin,
                                                       float* __restrict__ gmax,
                                                       int n, int rowsPerBlk) {
    int c  = threadIdx.x;                 // column 0..255 (lidar 0..127, cam 128..255)
    int r0 = blockIdx.x * rowsPerBlk;
    int r1 = min(n, r0 + rowsPerBlk);
    if (r0 >= r1) return;
    float mn = __builtin_huge_valf();
    float mx = -__builtin_huge_valf();
    int cb = bidx[r0];
    for (int r = r0; r < r1; ++r) {
        int b = bidx[r];
        if (b != cb) {
            atomicMinF(&gmin[cb * XCOLS + c], mn);
            atomicMaxF(&gmax[cb * XCOLS + c], mx);
            mn = __builtin_huge_valf();
            mx = -__builtin_huge_valf();
            cb = b;
        }
        float v = (c < LCOLS) ? lidar[r * LCOLS + c] : cam[r * CCOLS + (c - LCOLS)];
        mn = fminf(mn, v);
        mx = fmaxf(mx, v);
    }
    atomicMinF(&gmin[cb * XCOLS + c], mn);
    atomicMaxF(&gmax[cb * XCOLS + c], mx);
}

// ---------------------------------------------------- attention MLPs ------
// att[b*256 + 0..127]  = att_l[b], att[b*256 + 128..255] = att_c[b]
__global__ __launch_bounds__(256) void CMAFM_att(const float* __restrict__ gmin,
                                                 const float* __restrict__ gmax,
                                                 const float* __restrict__ Wl1,
                                                 const float* __restrict__ Wl2,
                                                 const float* __restrict__ Wc1,
                                                 const float* __restrict__ Wc2,
                                                 float* __restrict__ att) {
    __shared__ float cat[CADIM];
    __shared__ float h[HDIM + 2];
    int mlp = blockIdx.x >> 3;            // 0 = lidar MLP, 1 = cam MLP
    int b   = blockIdx.x & 7;
    const float* W1 = mlp ? Wc1 : Wl1;    // [512,170]
    const float* W2 = mlp ? Wc2 : Wl2;    // [170,128]
    int t = threadIdx.x;
    for (int i = t; i < CADIM; i += 256) {
        float v;
        if (i < 128)       v = gmin[b * XCOLS + i];            // seg_min(lidar)
        else if (i < 256)  v = gmax[b * XCOLS + (i - 128)];    // seg_max(lidar)
        else if (i < 384)  v = gmin[b * XCOLS + 128 + (i - 256)]; // seg_min(cam)
        else               v = gmax[b * XCOLS + 128 + (i - 384)]; // seg_max(cam)
        cat[i] = v;
    }
    __syncthreads();
    if (t < HDIM) {
        float s = 0.f;
        for (int k = 0; k < CADIM; ++k) s += cat[k] * W1[k * HDIM + t];
        h[t] = fmaxf(s, 0.f);
    }
    __syncthreads();
    if (t < 128) {
        float s = 0.f;
        for (int j = 0; j < HDIM; ++j) s += h[j] * W2[j * 128 + t];
        float r = fmaxf(s, 0.f);
        att[b * XCOLS + mlp * 128 + t] = 1.f / (1.f + expf(-r));
    }
}

// ------------------------------------------- fused gated double GEMM ------
// LDS: Wt1 f16 [256][256] | Wt2 f16 [256][256] | y f16 [8 waves][16][256]
__global__ __launch_bounds__(256) void CMAFM_fuse(const float* __restrict__ lidar,
                                                  const float* __restrict__ cam,
                                                  const int* __restrict__ bidx,
                                                  const float* __restrict__ att,
                                                  const _Float16* __restrict__ wt,
                                                  float* __restrict__ out, int n) {
    extern __shared__ char smem[];
    _Float16* sW1 = (_Float16*)smem;          // 65536 halves
    _Float16* sW2 = sW1 + 65536;              // 65536 halves
    _Float16* sY  = sW2 + 65536;              // 8 * 4096 halves

    // stage both weight matrices (f16, K-contiguous) into LDS, coalesced b128
    {
        const v4f* src = (const v4f*)wt;      // 131072 halves = 16384 x 16B
        v4f* dst = (v4f*)sW1;
        for (int i = threadIdx.x; i < 16384; i += 256) dst[i] = src[i];
    }
    __syncthreads();

    int wave = threadIdx.x >> 5;
    int lane = threadIdx.x & 31;
    int halfSel = lane >> 4;                  // 0: lanes 0-15, 1: lanes 16-31
    int lr = lane & 15;
    int rowb = (blockIdx.x * WAVES_PER_BLK + wave) * 16;

    int arow = min(rowb + lr, n - 1);         // clamp tail (loads only)
    int b = bidx[arow];
    const float* attb = att + b * XCOLS;
    const float* lrow = lidar + arow * LCOLS;
    const float* crow = cam + arow * CCOLS;

    // Build gated-x A fragments, register resident: x read from HBM once.
    // A 16x32 f16 layout: lane<16 -> row lr, K [k0..k0+8) and [k0+16..k0+24);
    // lane>=16 -> row lr, K offset +8 and +24.
    v16h Ax[8];
#pragma unroll
    for (int ks = 0; ks < 8; ++ks) {
        int k0 = ks * 32 + halfSel * 8;
        union { v16h v; _Float16 hh[16]; } u;
#pragma unroll
        for (int g = 0; g < 2; ++g) {
            int c0 = k0 + g * 16;
            const float* src = (c0 < LCOLS) ? (lrow + c0) : (crow + (c0 - LCOLS));
            const float* ap = attb + c0;
#pragma unroll
            for (int j = 0; j < 8; ++j)
                u.hh[g * 8 + j] = (_Float16)(src[j] * ap[j]);
        }
        Ax[ks] = u.v;
    }

    _Float16* yw = sY + wave * 4096;          // this wave's 16x256 y tile

    // GEMM1: y = relu(x @ Wf1)
    for (int t = 0; t < 16; ++t) {
        v8f acc = {};
#pragma unroll
        for (int ks = 0; ks < 8; ++ks) {
            union { v16h v; v4f f[2]; } ub;
            const _Float16* wp = sW1 + (t * 16 + lr) * 256 + ks * 32 + halfSel * 8;
            ub.f[0] = *(const v4f*)wp;
            ub.f[1] = *(const v4f*)(wp + 16);
            acc = __builtin_amdgcn_wmma_f32_16x16x32_f16(false, Ax[ks], false, ub.v,
                                                         (short)0, acc, false, false);
        }
#pragma unroll
        for (int v = 0; v < 8; ++v) {
            int m = halfSel * 8 + v;          // C layout: vgpr v -> row v / 8+v
            yw[m * 256 + t * 16 + lr] = (_Float16)fmaxf(acc[v], 0.f);
        }
    }
    __syncthreads();                          // uniform; also orders LDS within wave

    // Reload y as A fragments (row-major LDS -> A layout via b128 ds loads)
    v16h Ay[8];
#pragma unroll
    for (int ks = 0; ks < 8; ++ks) {
        union { v16h v; v4f f[2]; } u;
        const _Float16* yp = yw + lr * 256 + ks * 32 + halfSel * 8;
        u.f[0] = *(const v4f*)yp;
        u.f[1] = *(const v4f*)(yp + 16);
        Ay[ks] = u.v;
    }

    // GEMM2: out = relu(y @ Wf2)
    for (int t = 0; t < 16; ++t) {
        v8f acc = {};
#pragma unroll
        for (int ks = 0; ks < 8; ++ks) {
            union { v16h v; v4f f[2]; } ub;
            const _Float16* wp = sW2 + (t * 16 + lr) * 256 + ks * 32 + halfSel * 8;
            ub.f[0] = *(const v4f*)wp;
            ub.f[1] = *(const v4f*)(wp + 16);
            acc = __builtin_amdgcn_wmma_f32_16x16x32_f16(false, Ay[ks], false, ub.v,
                                                         (short)0, acc, false, false);
        }
#pragma unroll
        for (int v = 0; v < 8; ++v) {
            int m = rowb + halfSel * 8 + v;
            if (m < n) out[m * OUTC + t * 16 + lr] = fmaxf(acc[v], 0.f);
        }
    }
}

// ---------------------------------------------------------------- launch --
extern "C" void kernel_launch(void* const* d_in, const int* in_sizes, int n_in,
                              void* d_out, int out_size, void* d_ws, size_t ws_size,
                              hipStream_t stream) {
    const float* lidar = (const float*)d_in[0];
    const float* cam   = (const float*)d_in[1];
    const int*   bidx  = (const int*)d_in[2];
    const float* Wl1   = (const float*)d_in[3];
    const float* Wl2   = (const float*)d_in[4];
    const float* Wc1   = (const float*)d_in[5];
    const float* Wc2   = (const float*)d_in[6];
    const float* Wf1   = (const float*)d_in[7];
    const float* Wf2   = (const float*)d_in[8];
    float* out = (float*)d_out;
    int n = in_sizes[2];                       // voxel count N

    // workspace layout
    float* gmin = (float*)d_ws;                // [8][256]
    float* gmax = gmin + NB * XCOLS;           // [8][256]
    float* att  = gmax + NB * XCOLS;           // [8][256]
    _Float16* wt = (_Float16*)(att + NB * XCOLS); // 131072 halves (Wt1 | Wt2)

    CMAFM_prep<<<(131072 + 4096 + 255) / 256, 256, 0, stream>>>(Wf1, Wf2, wt, gmin, gmax);

    int gridA = 400;
    int rowsPerBlk = (n + gridA - 1) / gridA;
    CMAFM_segminmax<<<gridA, 256, 0, stream>>>(lidar, cam, bidx, gmin, gmax, n, rowsPerBlk);

    CMAFM_att<<<16, 256, 0, stream>>>(gmin, gmax, Wl1, Wl2, Wc1, Wc2, att);

    int blocksC = (n + ROWS_PER_BLK - 1) / ROWS_PER_BLK;
    size_t ldsBytes = (size_t)(65536 + 65536 + WAVES_PER_BLK * 4096) * sizeof(_Float16); // 320 KB
    CMAFM_fuse<<<blocksC, 256, ldsBytes, stream>>>(lidar, cam, bidx, att, wt, out, n);
}
